// BTokenizer_10058813407437
// MI455X (gfx1250) — matrix operand order repaired and checked by reading, not compiled, not run on hardware
//
#include <hip/hip_runtime.h>

// ---------------------------------------------------------------------------
// MI455X (gfx1250) residual-MLP:
//   y   = x @ Win^T + bin                       [16384,1024]
//   6x: y += GELU( LN(y) @ Wb[i]^T + bb[i] )
//   out = y @ Wout^T + bout                     [16384,2048]
// GEMMs: bf16 WMMA (v_wmma_f32_16x16x32_bf16), f32 accumulate.
// Tiles are DMA'd into LDS by the Tensor Data Mover (tensor_load_to_lds,
// TENSORcnt), double-buffered so TDM overlaps WMMA issue. LN/GELU stay f32.
// Working set (y f32 64MB + act bf16 32MB + W bf16 18MB) is L2-resident.
// ---------------------------------------------------------------------------

typedef __attribute__((ext_vector_type(16))) __bf16 v16bf;
typedef __attribute__((ext_vector_type(8)))  float  v8f;
typedef __attribute__((ext_vector_type(4)))  unsigned int u32x4;
typedef __attribute__((ext_vector_type(8)))  int i32x8;
typedef __attribute__((ext_vector_type(4)))  int i32x4;

#define N_TOKENS 16384
#define NIN      1024
#define HID      1024
#define NOUT     2048
#define NB_INNER 6

#if __has_builtin(__builtin_amdgcn_tensor_load_to_lds) && \
    __has_builtin(__builtin_amdgcn_s_wait_tensorcnt)
#define USE_TDM 1
#else
#define USE_TDM 0
#endif

union Frag {
    v16bf v;
    uint4 q[2];
};

__device__ __forceinline__ float gelu_exact(float x) {
    return 0.5f * x * (1.0f + erff(x * 0.70710678118654752f));
}

// ------------------------- f32 -> bf16 convert -----------------------------
__global__ void cvt_f32_bf16(const float* __restrict__ in,
                             __bf16* __restrict__ out, int n) {
    int i = blockIdx.x * blockDim.x + threadIdx.x;
    int stride = gridDim.x * blockDim.x;
    for (; i < n; i += stride) out[i] = (__bf16)in[i];
}

// --------------------- LayerNorm (wave per row) ----------------------------
__global__ void ln_rows(const float* __restrict__ y, __bf16* __restrict__ a) {
    const int row  = blockIdx.x * 8 + (threadIdx.x >> 5);
    const int lane = threadIdx.x & 31;
    const float* yr = y + (size_t)row * HID;
    float vals[HID / 32];
    float s = 0.f, ss = 0.f;
#pragma unroll
    for (int i = 0; i < HID / 32; ++i) {
        float v = yr[lane + 32 * i];
        vals[i] = v;
        s += v;
        ss += v * v;
    }
#pragma unroll
    for (int off = 16; off > 0; off >>= 1) {
        s  += __shfl_xor(s,  off, 32);
        ss += __shfl_xor(ss, off, 32);
    }
    const float mu  = s * (1.0f / HID);
    const float var = ss * (1.0f / HID) - mu * mu;
    const float inv = rsqrtf(var + 1e-5f);
    __bf16* ar = a + (size_t)row * HID;
#pragma unroll
    for (int i = 0; i < HID / 32; ++i)
        ar[lane + 32 * i] = (__bf16)((vals[i] - mu) * inv);
}

// ----------------------- TDM tile load (D# build) --------------------------
// Loads a tile_rows x 32 (bf16) tile from a row-major [rows,K] matrix at
// (row0, k0) into LDS at byte offset lds_off. LDS rows are padded to 80B
// (64B data + 16B pad) via D# pad fields: pad_interval=3 (16 DWORDs),
// pad_amount=3 (4 DWORDs) -> identical layout to the ds_load addressing below.
#if USE_TDM
__device__ __forceinline__ void tdm_load_tile(unsigned lds_off,
                                              const __bf16* gbase,
                                              unsigned row0, unsigned k0,
                                              unsigned rows, unsigned K,
                                              unsigned tile_rows) {
    unsigned long long ga =
        (unsigned long long)gbase + 2ull * ((unsigned long long)row0 * K + k0);
    u32x4 g0;
    g0[0] = 1u;                                   // count=1 (valid), user mode
    g0[1] = lds_off;                              // lds_addr (bytes)
    g0[2] = (unsigned)(ga & 0xFFFFFFFFu);         // global_addr[31:0]
    g0[3] = (unsigned)((ga >> 32) & 0x1FFFFFFu)   // global_addr[56:32]
          | (2u << 30);                           // type=2 ("image")
    i32x8 g1;
    g1[0] = (int)((1u << 16)      // data_size=1 (2 bytes)
                | (1u << 20)      // pad_enable
                | (3u << 22)      // pad_interval: 16 DWORDs (64B)
                | (3u << 25));    // pad_amount:   4 DWORDs (16B)
    g1[1] = (int)((K & 0xFFFFu) << 16);                       // tensor_dim0 lo
    g1[2] = (int)(((K >> 16) & 0xFFFFu) | ((rows & 0xFFFFu) << 16));
    g1[3] = (int)(((rows >> 16) & 0xFFFFu) | (32u << 16));    // tile_dim0=32
    g1[4] = (int)(tile_rows & 0xFFFFu);                       // tile_dim1
    g1[5] = (int)K;                                           // dim0_stride lo
    g1[6] = 0;
    g1[7] = 0;
    i32x4 z4 = {0, 0, 0, 0};
    i32x8 z8 = {0, 0, 0, 0, 0, 0, 0, 0};
    // 6-arg toolchain form: (g0, g1, g2, g3, extra, cpol)
    __builtin_amdgcn_tensor_load_to_lds(g0, g1, z4, z4, z8, 0);
}
#endif

// ------------------------------ WMMA GEMM ----------------------------------
// C[m,n] = sum_k A[m,k]*W[n,k] + bias[n]   (A:[M,K] bf16, W:[N,K] bf16)
// MODE 0/2: Y = acc + bias   MODE 1: Y += gelu(acc + bias)
// Block tile 128x256, 8 waves (2x4), wave tile 64x64, K step 32,
// double-buffered LDS filled by the TDM.
template <int MODE>
__global__ __launch_bounds__(256)
void gemm_bf16_wmma(const __bf16* __restrict__ A, const __bf16* __restrict__ W,
                    const float* __restrict__ bias, float* __restrict__ Y,
                    int M, int N, int K) {
    constexpr int LDT = 40;                  // halfs per padded LDS row (80B)
    constexpr int A_HALFS = 128 * LDT;       // 5120
    constexpr int W_HALFS = 256 * LDT;       // 10240
    constexpr int BUF_HALFS = A_HALFS + W_HALFS;
    __shared__ __bf16 smem[2 * BUF_HALFS];   // [A0 W0][A1 W1] = 60KB

    const int tid  = threadIdx.x;
    const int lane = tid & 31;
    const int wid  = tid >> 5;
    const int wm   = wid >> 2;   // 0..1 (64 rows each)
    const int wn   = wid & 3;    // 0..3 (64 cols each)
    const int bm   = blockIdx.y * 128;
    const int bn   = blockIdx.x * 256;
    const int hi   = lane >> 4;
    const int lr   = lane & 15;

    v8f acc[4][4];
#pragma unroll
    for (int mt = 0; mt < 4; ++mt)
#pragma unroll
        for (int nt = 0; nt < 4; ++nt) acc[mt][nt] = (v8f)(0.0f);

#if USE_TDM
    // prologue: DMA first K-slab into buffer 0 (wave 0 only issues)
    if (wid == 0) {
        tdm_load_tile(0u, A, (unsigned)bm, 0u, (unsigned)M, (unsigned)K, 128u);
        tdm_load_tile((unsigned)(A_HALFS * 2), W, (unsigned)bn, 0u,
                      (unsigned)N, (unsigned)K, 256u);
    }
#endif

    for (int k0 = 0; k0 < K; k0 += 32) {
        const int cur = (k0 >> 5) & 1;
        const __bf16* sA = smem + cur * BUF_HALFS;
        const __bf16* sW = smem + cur * BUF_HALFS + A_HALFS;
#if USE_TDM
        const bool has_next = (k0 + 32) < K;
        if (wid == 0) {
            if (has_next) {
                const int nxt = cur ^ 1;
                tdm_load_tile((unsigned)(nxt * BUF_HALFS * 2), A, (unsigned)bm,
                              (unsigned)(k0 + 32), (unsigned)M, (unsigned)K,
                              128u);
                tdm_load_tile((unsigned)((nxt * BUF_HALFS + A_HALFS) * 2), W,
                              (unsigned)bn, (unsigned)(k0 + 32), (unsigned)N,
                              (unsigned)K, 256u);
                __builtin_amdgcn_s_wait_tensorcnt((short)2);  // cur complete
            } else {
                __builtin_amdgcn_s_wait_tensorcnt((short)0);
            }
        }
        __syncthreads();   // publish cur buffer to all waves
#else
        // fallback staging: per-thread global->LDS copy of cur buffer
        {
            __bf16* dA = smem + cur * BUF_HALFS;
            __bf16* dW = smem + cur * BUF_HALFS + A_HALFS;
            const int lrow = tid >> 1;
            const int lcol = (tid & 1) * 16;
            const uint4* ga = reinterpret_cast<const uint4*>(
                A + (size_t)(bm + lrow) * K + k0 + lcol);
            uint4 a0 = ga[0], a1 = ga[1];
            const uint4* gw0 = reinterpret_cast<const uint4*>(
                W + (size_t)(bn + tid) * K + k0);
            uint4 w0 = gw0[0], w1 = gw0[1], w2 = gw0[2], w3 = gw0[3];
            __syncthreads();
            *reinterpret_cast<uint4*>(&dA[lrow * LDT + lcol])     = a0;
            *reinterpret_cast<uint4*>(&dA[lrow * LDT + lcol + 8]) = a1;
            uint4* dwp = reinterpret_cast<uint4*>(&dW[tid * LDT]);
            dwp[0] = w0; dwp[1] = w1; dwp[2] = w2; dwp[3] = w3;
            __syncthreads();
        }
#endif
        // ---- fragment fetch (ISA VGPR layouts) ----
        Frag af[4];
#pragma unroll
        for (int mt = 0; mt < 4; ++mt) {
            const int r = wm * 64 + mt * 16 + lr;
            af[mt].q[0] = *reinterpret_cast<const uint4*>(&sA[r * LDT + hi * 8]);
            af[mt].q[1] =
                *reinterpret_cast<const uint4*>(&sA[r * LDT + hi * 8 + 16]);
        }
        Frag bfr[4];
#pragma unroll
        for (int nt = 0; nt < 4; ++nt) {
            const int c = wn * 64 + nt * 16 + lr;
            bfr[nt].q[0] =
                *reinterpret_cast<const uint4*>(&sW[c * LDT + hi * 8]);
            bfr[nt].q[1] =
                *reinterpret_cast<const uint4*>(&sW[c * LDT + hi * 8 + 16]);
        }
        // ---- 16 independent WMMAs per wave per K step ----
#pragma unroll
        for (int mt = 0; mt < 4; ++mt)
#pragma unroll
            for (int nt = 0; nt < 4; ++nt)
                acc[mt][nt] = __builtin_amdgcn_wmma_f32_16x16x32_bf16(
                    false, af[mt].v, false, bfr[nt].v, (short)0, acc[mt][nt],
                    false, false);
        __syncthreads();   // all waves done reading cur before it is refilled
    }

    // ---- epilogue: C layout -> lane holds (m = i + 8*hi, n = lane&15) ----
#pragma unroll
    for (int mt = 0; mt < 4; ++mt) {
#pragma unroll
        for (int nt = 0; nt < 4; ++nt) {
            const int col = bn + wn * 64 + nt * 16 + lr;
            const float b = bias[col];
            const int rbase = bm + wm * 64 + mt * 16 + hi * 8;
#pragma unroll
            for (int i = 0; i < 8; ++i) {
                const size_t idx = (size_t)(rbase + i) * N + col;
                const float v = acc[mt][nt][i] + b;
                if (MODE == 1)
                    Y[idx] += gelu_exact(v);
                else
                    Y[idx] = v;
            }
        }
    }
}

// ---------------------------------------------------------------------------
extern "C" void kernel_launch(void* const* d_in, const int* in_sizes, int n_in,
                              void* d_out, int out_size, void* d_ws, size_t ws_size,
                              hipStream_t stream) {
    const float* x    = (const float*)d_in[0];
    const float* Win  = (const float*)d_in[1];
    const float* binb = (const float*)d_in[2];
    const float* Wb   = (const float*)d_in[3];
    const float* bb   = (const float*)d_in[4];
    const float* Wout = (const float*)d_in[5];
    const float* bout = (const float*)d_in[6];
    float* out = (float*)d_out;

    char* ws = (char*)d_ws;
    float*  ybuf = (float*)ws;                                 // 64MB f32 y
    __bf16* abuf = (__bf16*)(ws + (size_t)N_TOKENS * HID * 4); // 32MB bf16 act
    __bf16* wbuf = (__bf16*)(ws + (size_t)N_TOKENS * HID * 6); // 18MB bf16 W
    __bf16* win_bf  = wbuf;
    __bf16* wb_bf   = win_bf + (size_t)HID * NIN;
    __bf16* wout_bf = wb_bf + (size_t)NB_INNER * HID * HID;

    const int CT = 256, CB = 2048;
    cvt_f32_bf16<<<CB, CT, 0, stream>>>(x, abuf, N_TOKENS * NIN);
    cvt_f32_bf16<<<CB, CT, 0, stream>>>(Win, win_bf, HID * NIN);
    cvt_f32_bf16<<<CB, CT, 0, stream>>>(Wb, wb_bf, NB_INNER * HID * HID);
    cvt_f32_bf16<<<CB, CT, 0, stream>>>(Wout, wout_bf, NOUT * HID);

    const dim3 blk(256);
    const dim3 grid_h(HID / 256, N_TOKENS / 128);    // 4 x 128
    const dim3 grid_o(NOUT / 256, N_TOKENS / 128);   // 8 x 128

    gemm_bf16_wmma<0><<<grid_h, blk, 0, stream>>>(abuf, win_bf, binb, ybuf,
                                                  N_TOKENS, HID, NIN);
    for (int i = 0; i < NB_INNER; ++i) {
        ln_rows<<<N_TOKENS / 8, 256, 0, stream>>>(ybuf, abuf);
        gemm_bf16_wmma<1><<<grid_h, blk, 0, stream>>>(
            abuf, wb_bf + (size_t)i * HID * HID, bb + (size_t)i * HID, ybuf,
            N_TOKENS, HID, HID);
    }
    cvt_f32_bf16<<<CB, CT, 0, stream>>>(ybuf, abuf, N_TOKENS * HID);
    gemm_bf16_wmma<2><<<grid_o, blk, 0, stream>>>(abuf, wout_bf, bout, out,
                                                  N_TOKENS, NOUT, HID);
}
